// MultiHeadAttention_60241211293738
// MI455X (gfx1250) — compile-verified, matrix operand order checked by reading, not run on hardware
//
#include <hip/hip_runtime.h>
#include <math.h>

// ---------------------------------------------------------------------------
// MHA for MI455X (gfx1250).  All GEMM stages on v_wmma_f32_16x16x32_bf16.
// fp32->bf16 conversion hoisted out of hot loops; LDS staging uses
// GLOBAL_LOAD_ASYNC_TO_LDS_B128 (ASYNCcnt) with double buffering; softmax
// cross-lane reductions on v_permlane16 (VALU) instead of ds_bpermute.
// V is produced pre-transposed [bh][d][s] so attention stages it async too.
// bs=4, seq=2048, d_model=1024, heads=16, head_dim=64.
// ---------------------------------------------------------------------------

typedef __bf16 bf16;
typedef __attribute__((ext_vector_type(16))) __bf16 v16bf;
typedef __attribute__((ext_vector_type(8)))  __bf16 v8bf;
typedef __attribute__((ext_vector_type(8)))  float  v8f;
typedef __attribute__((ext_vector_type(8)))  unsigned short v8u;
typedef __attribute__((ext_vector_type(4)))  int    v4i;

#define D_MODEL 1024
#define SEQ     2048
#define HEADS   16
#define HDIM    64
#define TOKENS  8192   // 4 * 2048

__device__ __forceinline__ unsigned short f2bfu(float f) {
  unsigned int u = __builtin_bit_cast(unsigned int, f);
  unsigned int r = u + 0x7fffu + ((u >> 16) & 1u);   // round-to-nearest-even
  return (unsigned short)(r >> 16);
}

// ------------------------- async copy plumbing -----------------------------
#if __has_builtin(__builtin_amdgcn_global_load_async_to_lds_b128)
#define CDNA5_HAS_ASYNC 1
#else
#define CDNA5_HAS_ASYNC 0
#endif

__device__ __forceinline__ void async_cp16(const void* g, void* l) {
#if CDNA5_HAS_ASYNC
  __builtin_amdgcn_global_load_async_to_lds_b128(
      (__attribute__((address_space(1))) v4i*)g,
      (__attribute__((address_space(3))) v4i*)l, 0, 0);
#else
  *(v8u*)l = *(const v8u*)g;     // synchronous fallback
#endif
}

#if CDNA5_HAS_ASYNC
#if __has_builtin(__builtin_amdgcn_s_wait_asynccnt)
#define ASYNC_WAIT(n) __builtin_amdgcn_s_wait_asynccnt(n)
#else
#define ASYNC_WAIT(n) asm volatile("s_wait_asynccnt " #n ::: "memory")
#endif
#else
#define ASYNC_WAIT(n) do {} while (0)
#endif

// ---------------- 16-lane-row xor reductions on the VALU pipe --------------
#if __has_builtin(__builtin_amdgcn_permlane16)
__device__ __forceinline__ float lanexor16(float x, unsigned lo, unsigned hi) {
  unsigned u = __builtin_bit_cast(unsigned, x);
  unsigned r = __builtin_amdgcn_permlane16(u, u, lo, hi, false, false);
  return __builtin_bit_cast(float, r);
}
__device__ __forceinline__ float redmax16(float x) {
  x = fmaxf(x, lanexor16(x, 0x67452301u, 0xEFCDAB89u));   // xor 1
  x = fmaxf(x, lanexor16(x, 0x54761032u, 0xDCFE98BAu));   // xor 2
  x = fmaxf(x, lanexor16(x, 0x32107654u, 0xBA98FEDCu));   // xor 4
  x = fmaxf(x, lanexor16(x, 0xFEDCBA98u, 0x76543210u));   // xor 8
  return x;
}
__device__ __forceinline__ float redsum16(float x) {
  x += lanexor16(x, 0x67452301u, 0xEFCDAB89u);
  x += lanexor16(x, 0x54761032u, 0xDCFE98BAu);
  x += lanexor16(x, 0x32107654u, 0xBA98FEDCu);
  x += lanexor16(x, 0xFEDCBA98u, 0x76543210u);
  return x;
}
#else
__device__ __forceinline__ float redmax16(float x) {
  x = fmaxf(x, __shfl_xor(x, 1, 16));
  x = fmaxf(x, __shfl_xor(x, 2, 16));
  x = fmaxf(x, __shfl_xor(x, 4, 16));
  x = fmaxf(x, __shfl_xor(x, 8, 16));
  return x;
}
__device__ __forceinline__ float redsum16(float x) {
  x += __shfl_xor(x, 1, 16);
  x += __shfl_xor(x, 2, 16);
  x += __shfl_xor(x, 4, 16);
  x += __shfl_xor(x, 8, 16);
  return x;
}
#endif

// Build a 16-bit A/B WMMA fragment from two contiguous 8-element (16B) LDS
// runs: elements 0..7 = k..k+7, elements 8..15 = k+16..k+23 (ISA 7.12.2).
__device__ __forceinline__ v16bf lds_frag(const unsigned short* p0,
                                          const unsigned short* p1) {
  v8bf lo = *reinterpret_cast<const v8bf*>(p0);
  v8bf hi = *reinterpret_cast<const v8bf*>(p1);
  return __builtin_shufflevector(lo, hi, 0,1,2,3,4,5,6,7,8,9,10,11,12,13,14,15);
}

// ---------------------------------------------------------------------------
// One-time converters (bandwidth-bound, run once each).
// ---------------------------------------------------------------------------
__global__ __launch_bounds__(256)
void cvt_bf16(const float* __restrict__ in, unsigned short* __restrict__ out) {
  int i = (blockIdx.x * 256 + threadIdx.x) * 8;
  float4 a = *reinterpret_cast<const float4*>(in + i);
  float4 b = *reinterpret_cast<const float4*>(in + i + 4);
  v8u o;
  o[0] = f2bfu(a.x); o[1] = f2bfu(a.y); o[2] = f2bfu(a.z); o[3] = f2bfu(a.w);
  o[4] = f2bfu(b.x); o[5] = f2bfu(b.y); o[6] = f2bfu(b.z); o[7] = f2bfu(b.w);
  *reinterpret_cast<v8u*>(out + i) = o;
}

// W[k][n] fp32 -> Wt[n][k] bf16, 64x64 LDS tile.
__global__ __launch_bounds__(256)
void cvt_tr_bf16(const float* __restrict__ W, unsigned short* __restrict__ Wt) {
  __shared__ unsigned short t[64 * 65];
  const int tid = threadIdx.x;
  const int k0 = blockIdx.y * 64, n0 = blockIdx.x * 64;
  #pragma unroll
  for (int i = 0; i < 4; ++i) {
    int v = tid + 256 * i;
    int r = v >> 4, c4 = (v & 15) * 4;                 // k-row, n-col
    float4 x = *reinterpret_cast<const float4*>(W + (size_t)(k0 + r) * D_MODEL + n0 + c4);
    t[(c4 + 0) * 65 + r] = f2bfu(x.x);
    t[(c4 + 1) * 65 + r] = f2bfu(x.y);
    t[(c4 + 2) * 65 + r] = f2bfu(x.z);
    t[(c4 + 3) * 65 + r] = f2bfu(x.w);
  }
  __syncthreads();
  #pragma unroll
  for (int i = 0; i < 4; ++i) {
    int v = tid + 256 * i;
    int n = v >> 4, k4 = (v & 15) * 4;
    ushort4 o;
    o.x = t[n * 65 + k4 + 0]; o.y = t[n * 65 + k4 + 1];
    o.z = t[n * 65 + k4 + 2]; o.w = t[n * 65 + k4 + 3];
    *reinterpret_cast<ushort4*>(Wt + (size_t)(n0 + n) * D_MODEL + k0 + k4) = o;
  }
}

// ---------------------------------------------------------------------------
// GEMM: C[8192 x 1024] = A[8192 x 1024] * Bt^T + bias.  A bf16 [m][k],
// Bt bf16 [n][k] (pre-transposed).  Block tile 128x128, 8 waves, wave tile
// 32x64.  Double-buffered async LDS staging.
// MODE==0: fp32 row-major.  MODE==1: bf16 scatter to [b,h,s,d] (Q,K).
// MODE==2: bf16 transposed [b,h,d,s] with packed b128 stores (V).
// ---------------------------------------------------------------------------
template <int MODE>
__global__ __launch_bounds__(256)
void gemm_bf16(const unsigned short* __restrict__ A,
               const unsigned short* __restrict__ Bt,
               const float* __restrict__ bias, void* __restrict__ Cv) {
  __shared__ unsigned short sA[2][128 * 32];   // [m][k]
  __shared__ unsigned short sB[2][128 * 32];   // [n][k]

  const int tid  = threadIdx.x;
  const int lane = tid & 31;
  const int w    = tid >> 5;
  const int wm   = w & 3;
  const int wn   = w >> 2;
  const int l16  = lane & 15;
  const int hgrp = lane >> 4;
  const int kb   = hgrp * 8;
  const int m0   = blockIdx.y * 128;
  const int n0   = blockIdx.x * 128;

  v8f acc[2][4];
  #pragma unroll
  for (int ms = 0; ms < 2; ++ms)
    #pragma unroll
    for (int ns = 0; ns < 4; ++ns) {
      float bv_ = bias[n0 + wn * 64 + ns * 16 + l16];
      #pragma unroll
      for (int i = 0; i < 8; ++i) acc[ms][ns][i] = bv_;
    }

  // 4 async 16B chunks per thread per stage (2 for A tile, 2 for B tile).
  auto stage = [&](int kt, int buf) {
    #pragma unroll
    for (int i = 0; i < 2; ++i) {
      int c   = tid + 256 * i;
      int row = c >> 2, k8 = (c & 3) * 8;
      async_cp16(A  + (size_t)(m0 + row) * D_MODEL + kt + k8,
                 &sA[buf][row * 32 + k8]);
      async_cp16(Bt + (size_t)(n0 + row) * D_MODEL + kt + k8,
                 &sB[buf][row * 32 + k8]);
    }
  };

  stage(0, 0);
  int buf = 0;
  for (int kt = 0; kt < D_MODEL; kt += 32) {
    if (kt + 32 < D_MODEL) {
      stage(kt + 32, buf ^ 1);  // prefetch next tile, overlap with compute
      ASYNC_WAIT(4);            // current tile (4 oldest ops) landed
    } else {
      ASYNC_WAIT(0);            // tail: no prefetch in flight behind us
    }
    __syncthreads();

    v16bf af[2], bfr[4];
    #pragma unroll
    for (int ms = 0; ms < 2; ++ms) {
      int row = wm * 32 + ms * 16 + l16;
      af[ms] = lds_frag(&sA[buf][row * 32 + kb], &sA[buf][row * 32 + kb + 16]);
    }
    #pragma unroll
    for (int ns = 0; ns < 4; ++ns) {
      int n = wn * 64 + ns * 16 + l16;
      bfr[ns] = lds_frag(&sB[buf][n * 32 + kb], &sB[buf][n * 32 + kb + 16]);
    }
    #pragma unroll
    for (int ms = 0; ms < 2; ++ms)
      #pragma unroll
      for (int ns = 0; ns < 4; ++ns)
        acc[ms][ns] = __builtin_amdgcn_wmma_f32_16x16x32_bf16(
            false, af[ms], false, bfr[ns], (short)0, acc[ms][ns], false, false);

    __syncthreads();   // free 'buf' before next iteration's stage overwrites it
    buf ^= 1;
  }

  #pragma unroll
  for (int ms = 0; ms < 2; ++ms)
    #pragma unroll
    for (int ns = 0; ns < 4; ++ns) {
      int Ng = n0 + wn * 64 + ns * 16 + l16;
      if (MODE == 2) {
        // bf16 to [b,h,d,s]; the 8 accumulator rows are contiguous in s.
        int s0_ = (m0 & 2047) + wm * 32 + ms * 16 + 8 * hgrp;
        int bb  = m0 >> 11;
        int hh  = Ng >> 6, dd = Ng & 63;
        size_t idx = (((size_t)bb * HEADS + hh) * HDIM + dd) * SEQ + s0_;
        v8u o;
        #pragma unroll
        for (int i = 0; i < 8; ++i) o[i] = f2bfu(acc[ms][ns][i]);
        *reinterpret_cast<v8u*>((unsigned short*)Cv + idx) = o;
      } else {
        #pragma unroll
        for (int i = 0; i < 8; ++i) {
          int Mg = m0 + wm * 32 + ms * 16 + i + 8 * hgrp;
          if (MODE == 1) {     // bf16 scatter to [b,h,s,d]
            int bb = Mg >> 11, ss = Mg & 2047;
            int hh = Ng >> 6,  dd = Ng & 63;
            size_t idx = (((size_t)bb * HEADS + hh) * SEQ + ss) * HDIM + dd;
            ((unsigned short*)Cv)[idx] = f2bfu(acc[ms][ns][i]);
          } else {             // fp32 row-major (final output)
            ((float*)Cv)[(size_t)Mg * D_MODEL + Ng] = acc[ms][ns][i];
          }
        }
      }
    }
}

// ---------------------------------------------------------------------------
// Flash attention per (b,h).  Q/K bf16 [bh][s][64], V bf16 [bh][64][s].
// Wave = 16 queries, key blocks of 32: 4 WMMA scores + 4 WMMA PV.
// scale = 1/32 (reference's head_dim/2 quirk).
// Output bf16 [b, s, h*64+d] row-major for the output projection.
// ---------------------------------------------------------------------------
__global__ __launch_bounds__(256)
void attn_kernel(const unsigned short* __restrict__ Q,
                 const unsigned short* __restrict__ K,
                 const unsigned short* __restrict__ VT,
                 unsigned short* __restrict__ O) {
  __shared__ unsigned short sK [32 * 64];      // [key][d]
  __shared__ unsigned short sVT[64 * 32];      // [d][key]
  __shared__ unsigned short sP [8 * 16 * 32];  // per-wave P [m][key]

  const int tid  = threadIdx.x;
  const int lane = tid & 31;
  const int w    = tid >> 5;
  const int l16  = lane & 15;
  const int hgrp = lane >> 4;
  const int kb   = hgrp * 8;
  const int bh   = blockIdx.y;
  const int b    = bh >> 4, h = bh & 15;
  const size_t base = (size_t)bh * SEQ * HDIM;   // same footprint both layouts
  const int qr0  = blockIdx.x * 128 + w * 16;
  unsigned short* sPw = &sP[w * 512];

  // Q A-fragments straight from global bf16 (16B-aligned runs).
  v16bf qa[2];
  {
    const unsigned short* qrow = Q + base + (size_t)(qr0 + l16) * HDIM;
    #pragma unroll
    for (int f = 0; f < 2; ++f) {
      v8bf lo = *reinterpret_cast<const v8bf*>(qrow + f * 32 + kb);
      v8bf hi = *reinterpret_cast<const v8bf*>(qrow + f * 32 + kb + 16);
      qa[f] = __builtin_shufflevector(lo, hi,
              0,1,2,3,4,5,6,7,8,9,10,11,12,13,14,15);
    }
  }

  float m_run[8], l_run[8];
  #pragma unroll
  for (int i = 0; i < 8; ++i) { m_run[i] = -3.0e38f; l_run[i] = 0.0f; }
  v8f acc[4];
  #pragma unroll
  for (int t = 0; t < 4; ++t)
    #pragma unroll
    for (int i = 0; i < 8; ++i) acc[t][i] = 0.0f;

  for (int kb0 = 0; kb0 < SEQ; kb0 += 32) {
    __syncthreads();
    {
      // K tile [key][d]: 32x64 bf16 = 256 x 16B async chunks (1/thread).
      int key = tid >> 3, d8 = (tid & 7) * 8;
      async_cp16(K + base + (size_t)(kb0 + key) * HDIM + d8,
                 &sK[key * 64 + d8]);
      // V tile [d][key]: V is pre-transposed, also a pure async copy.
      int d = tid >> 2, k8 = (tid & 3) * 8;
      async_cp16(VT + base + (size_t)d * SEQ + kb0 + k8,
                 &sVT[d * 32 + k8]);
    }
    ASYNC_WAIT(0);
    __syncthreads();

    // Scores: S[16q x 32k] = Q(16x64) * K^T, two 16x16 C tiles.
    v8f s0 = {}; v8f s1 = {};
    {
      int n = l16;
      v16bf b00 = lds_frag(&sK[n * 64 + kb],      &sK[n * 64 + kb + 16]);
      v16bf b01 = lds_frag(&sK[n * 64 + 32 + kb], &sK[n * 64 + 32 + kb + 16]);
      s0 = __builtin_amdgcn_wmma_f32_16x16x32_bf16(false, qa[0], false, b00,
                                                   (short)0, s0, false, false);
      s0 = __builtin_amdgcn_wmma_f32_16x16x32_bf16(false, qa[1], false, b01,
                                                   (short)0, s0, false, false);
      int n1 = 16 + l16;
      v16bf b10 = lds_frag(&sK[n1 * 64 + kb],      &sK[n1 * 64 + kb + 16]);
      v16bf b11 = lds_frag(&sK[n1 * 64 + 32 + kb], &sK[n1 * 64 + 32 + kb + 16]);
      s1 = __builtin_amdgcn_wmma_f32_16x16x32_bf16(false, qa[0], false, b10,
                                                   (short)0, s1, false, false);
      s1 = __builtin_amdgcn_wmma_f32_16x16x32_bf16(false, qa[1], false, b11,
                                                   (short)0, s1, false, false);
    }
    s0 *= 0.03125f;   // 1/(head_dim/2)
    s1 *= 0.03125f;

    // Online softmax: row r = i + 8*hgrp lives in VGPR i across 16 lanes.
    #pragma unroll
    for (int i = 0; i < 8; ++i) {
      float mx = redmax16(fmaxf(s0[i], s1[i]));
      float mn = fmaxf(m_run[i], mx);
      float a  = __expf(m_run[i] - mn);
      m_run[i] = mn;
      float e0 = __expf(s0[i] - mn);
      float e1 = __expf(s1[i] - mn);
      float rs = redsum16(e0 + e1);
      l_run[i] = l_run[i] * a + rs;
      int row = i + 8 * hgrp;
      sPw[row * 32 + l16]      = f2bfu(e0);   // keys 0..15
      sPw[row * 32 + 16 + l16] = f2bfu(e1);   // keys 16..31
      acc[0][i] *= a; acc[1][i] *= a; acc[2][i] *= a; acc[3][i] *= a;
    }

    // PV: O(16x64) += P(16x32) * V(32x64).  DS in-order within wave.
    v16bf pa = lds_frag(&sPw[l16 * 32 + kb], &sPw[l16 * 32 + kb + 16]);
    #pragma unroll
    for (int t = 0; t < 4; ++t) {
      int n = t * 16 + l16;
      v16bf vb = lds_frag(&sVT[n * 32 + kb], &sVT[n * 32 + kb + 16]);
      acc[t] = __builtin_amdgcn_wmma_f32_16x16x32_bf16(false, pa, false, vb,
                                                       (short)0, acc[t],
                                                       false, false);
    }
  }

  #pragma unroll
  for (int i = 0; i < 8; ++i) {
    float inv = 1.0f / l_run[i];
    int srow = qr0 + i + 8 * hgrp;
    size_t rowoff = ((size_t)b * SEQ + srow) * D_MODEL + h * HDIM;
    #pragma unroll
    for (int t = 0; t < 4; ++t)
      O[rowoff + t * 16 + l16] = f2bfu(acc[t][i] * inv);
  }
}

// ---------------------------------------------------------------------------
extern "C" void kernel_launch(void* const* d_in, const int* in_sizes, int n_in,
                              void* d_out, int out_size, void* d_ws,
                              size_t ws_size, hipStream_t stream) {
  const float* X  = (const float*)d_in[0];
  const float* Wq = (const float*)d_in[1];
  const float* bq = (const float*)d_in[2];
  const float* Wk = (const float*)d_in[3];
  const float* bk = (const float*)d_in[4];
  const float* Wv = (const float*)d_in[5];
  const float* bv = (const float*)d_in[6];
  const float* Wo = (const float*)d_in[7];
  const float* bo = (const float*)d_in[8];
  float* out = (float*)d_out;

  char* ws = (char*)d_ws;
  const size_t XB = (size_t)TOKENS * D_MODEL * 2;   // 16 MB (bf16 matrix)
  const size_t WB = (size_t)D_MODEL * D_MODEL * 2;  // 2 MB  (bf16 weight)
  unsigned short* Xb  = (unsigned short*)(ws);
  unsigned short* Wqt = (unsigned short*)(ws + XB);
  unsigned short* Wkt = (unsigned short*)(ws + XB + WB);
  unsigned short* Wvt = (unsigned short*)(ws + XB + 2 * WB);
  unsigned short* Wot = (unsigned short*)(ws + XB + 3 * WB);
  unsigned short* qb  = (unsigned short*)(ws + XB + 4 * WB);
  unsigned short* kbp = (unsigned short*)(ws + 2 * XB + 4 * WB);
  unsigned short* vtb = (unsigned short*)(ws + 3 * XB + 4 * WB);
  unsigned short* aob = (unsigned short*)(ws + 4 * XB + 4 * WB);

  dim3 blk(256, 1, 1);
  // One-time conversions.
  cvt_bf16<<<dim3(TOKENS * D_MODEL / (256 * 8), 1, 1), blk, 0, stream>>>(X, Xb);
  dim3 gt(D_MODEL / 64, D_MODEL / 64, 1);
  cvt_tr_bf16<<<gt, blk, 0, stream>>>(Wq, Wqt);
  cvt_tr_bf16<<<gt, blk, 0, stream>>>(Wk, Wkt);
  cvt_tr_bf16<<<gt, blk, 0, stream>>>(Wv, Wvt);
  cvt_tr_bf16<<<gt, blk, 0, stream>>>(Wo, Wot);

  dim3 gg(D_MODEL / 128, TOKENS / 128, 1);   // 8 x 64
  gemm_bf16<1><<<gg, blk, 0, stream>>>(Xb, Wqt, bq, qb);
  gemm_bf16<1><<<gg, blk, 0, stream>>>(Xb, Wkt, bk, kbp);
  gemm_bf16<2><<<gg, blk, 0, stream>>>(Xb, Wvt, bv, vtb);   // V pre-transposed
  attn_kernel<<<dim3(SEQ / 128, 4 * HEADS, 1), blk, 0, stream>>>(qb, kbp, vtb, aob);
  gemm_bf16<0><<<gg, blk, 0, stream>>>(aob, Wot, bo, out);
}
